// GRUBlock_20650202759651
// MI455X (gfx1250) — matrix operand order repaired and checked by reading, not compiled, not run on hardware
//
#include <hip/hip_runtime.h>
#include <hip/hip_bf16.h>
#include <math.h>

// Problem constants (match reference)
constexpr int Bv  = 16;     // batch
constexpr int Tv  = 4096;   // timesteps
constexpr int Dv  = 512;    // hidden = input = output width
constexpr int G3  = 3 * Dv; // 1536 gate rows
constexpr int HS  = 520;    // padded LDS row stride (bf16 elems) to avoid bank conflicts

typedef __bf16 v16bf __attribute__((ext_vector_type(16)));
typedef __bf16 v8bf  __attribute__((ext_vector_type(8)));
typedef float  v8f   __attribute__((ext_vector_type(8)));

static __device__ __forceinline__ v8f wmma_bf16(v16bf a, v16bf b, v8f c) {
    // D = A(16x32 bf16) x B(32x16 bf16) + C(16x16 f32)
    return __builtin_amdgcn_wmma_f32_16x16x32_bf16(
        /*neg_a=*/false, a, /*neg_b=*/false, b,
        /*c_mod=*/(short)0, c, /*reuse_a=*/false, /*reuse_b=*/false);
}

static __device__ __forceinline__ float gelu_exact(float x) {
    return 0.5f * x * (1.0f + erff(x * 0.7071067811865475f));
}

// A-fragment (16x32 bf16, MxK): lane L holds row m=L&15; with hi=L>>4,
// VGPR halves 0..7  hold K = kc*32 + hi*16 + 0..7
// VGPR halves 8..15 hold K = kc*32 + hi*16 + 8 + 0..7
template <bool GELU>
static __device__ __forceinline__ v16bf load_a_frag(const float* __restrict__ rowp,
                                                    int kc, int hi) {
    const float* p1 = rowp + kc * 32 + hi * 16;
    const float* p2 = p1 + 8;
    float4 x0 = *(const float4*)(p1);
    float4 x1 = *(const float4*)(p1 + 4);
    float4 x2 = *(const float4*)(p2);
    float4 x3 = *(const float4*)(p2 + 4);
    float f[16] = {x0.x, x0.y, x0.z, x0.w, x1.x, x1.y, x1.z, x1.w,
                   x2.x, x2.y, x2.z, x2.w, x3.x, x3.y, x3.z, x3.w};
    v16bf a;
#pragma unroll
    for (int i = 0; i < 16; ++i) {
        float v = f[i];
        if (GELU) v = gelu_exact(v);
        a[i] = (__bf16)v;
    }
    return a;
}

// B-fragment (32x16 bf16, KxN): lane L holds col n=L&15, K = kc*32 + (L>>4)*16 + 0..15
static __device__ __forceinline__ v16bf load_b_frag_global(const float* __restrict__ p) {
    float4 x0 = *(const float4*)(p);
    float4 x1 = *(const float4*)(p + 4);
    float4 x2 = *(const float4*)(p + 8);
    float4 x3 = *(const float4*)(p + 12);
    float f[16] = {x0.x, x0.y, x0.z, x0.w, x1.x, x1.y, x1.z, x1.w,
                   x2.x, x2.y, x2.z, x2.w, x3.x, x3.y, x3.z, x3.w};
    v16bf b;
#pragma unroll
    for (int i = 0; i < 16; ++i) b[i] = (__bf16)f[i];
    return b;
}

static __device__ __forceinline__ v16bf load_b_frag_lds(const __bf16* hp) {
    v8bf lo  = *(const v8bf*)(hp);
    v8bf hi8 = *(const v8bf*)(hp + 8);
    v16bf b;
#pragma unroll
    for (int i = 0; i < 8; ++i) { b[i] = lo[i]; b[8 + i] = hi8[i]; }
    return b;
}

// ---------------------------------------------------------------------------
// Input projection: out[r, n] = act(X)[r,:] . W[n,:] + bias[n]
// X: [R, 512] f32 (act = GELU for layer 2), W: [1536, 512] f32, out: [R, 1536] f32
// Grid: R/16 blocks, 256 threads (8 waves). Wave w covers 12 N-tiles (192 cols).
// ---------------------------------------------------------------------------
template <bool GELU>
__global__ __launch_bounds__(256, 1)
void xproj_gemm(const float* __restrict__ X, const float* __restrict__ W,
                const float* __restrict__ bias, float* __restrict__ out) {
    const int tid  = threadIdx.x;
    const int lane = tid & 31;
    const int wv   = tid >> 5;
    const int m    = lane & 15;
    const int hi   = lane >> 4;

    const size_t row = (size_t)blockIdx.x * 16 + m;
    const float* xr  = X + row * Dv;

    // Persistent A fragments for this 16-row block (K = 0..511)
    v16bf aw[16];
#pragma unroll
    for (int kc = 0; kc < 16; ++kc) aw[kc] = load_a_frag<GELU>(xr, kc, hi);

    const int rbase = blockIdx.x * 16 + 8 * hi;

#pragma unroll 1
    for (int nt = 0; nt < 12; ++nt) {
        const int n = wv * 192 + nt * 16 + m;   // output column (also W row)
        const float* wr = W + (size_t)n * Dv;
        const float bv  = bias[n];
        v8f acc0, acc1;
#pragma unroll
        for (int r = 0; r < 8; ++r) { acc0[r] = bv; acc1[r] = 0.f; }
        // two independent accumulation streams -> shorter WMMA RAW chains
#pragma unroll
        for (int kc = 0; kc < 16; kc += 2) {
            v16bf b0 = load_b_frag_global(wr + (kc + 0) * 32 + hi * 16);
            v16bf b1 = load_b_frag_global(wr + (kc + 1) * 32 + hi * 16);
            acc0 = wmma_bf16(aw[kc + 0], b0, acc0);
            acc1 = wmma_bf16(aw[kc + 1], b1, acc1);
        }
        v8f acc = acc0 + acc1;
#pragma unroll
        for (int r = 0; r < 8; ++r)
            out[(size_t)(rbase + r) * G3 + n] = acc[r];
    }
}

// ---------------------------------------------------------------------------
// Zero the h ping-pong buffers and the inter-WG barrier.
// ---------------------------------------------------------------------------
__global__ void init_state(float* __restrict__ hbuf, unsigned* __restrict__ bar) {
    int i = blockIdx.x * blockDim.x + threadIdx.x;
    if (i < 2 * Bv * Dv) hbuf[i] = 0.0f;
    if (i < 4) bar[i] = 0u;
}

// ---------------------------------------------------------------------------
// Persistent GRU scan: 16 workgroups x 192 threads (6 waves).
// WG b owns hidden units [32b, 32b+32); wave w (g=w>>1 gate, hf=w&1 half)
// owns 16 gate rows of W_hh kept as bf16 A-fragments in VGPRs for all T steps.
// Per step: gh^T = W_tile x h^T via 16 WMMAs/wave (2 accumulation streams),
// gates in f32, h broadcast through a ping-pong L2 buffer with one
// device-scope barrier per step.
// ---------------------------------------------------------------------------
__global__ __launch_bounds__(192, 1)
void gru_scan(const float* __restrict__ xp, const float* __restrict__ w_hh,
              const float* __restrict__ b_hh, float* __restrict__ y,
              float* __restrict__ hbuf, unsigned* __restrict__ bar) {
    __shared__ __bf16 h_bf[Bv * HS];          // broadcast h (bf16) for WMMA B
    __shared__ float  gh_s[3][32][17];        // gate pre-activations (padded)

    const int tid  = threadIdx.x;
    const int lane = tid & 31;
    const int wv   = tid >> 5;
    const int m    = lane & 15;               // A row within tile / D column (batch)
    const int hi   = lane >> 4;
    const int g    = wv >> 1;                 // gate: 0=r 1=z 2=n
    const int hf   = wv & 1;                  // which 16-unit half
    const int row_base = g * Dv + blockIdx.x * 32 + hf * 16;

    // --- one-time: W_hh tile -> bf16 A fragments in registers -------------
    const float* wr = w_hh + (size_t)(row_base + m) * Dv;
    v16bf aw[16];
#pragma unroll
    for (int kc = 0; kc < 16; ++kc) aw[kc] = load_a_frag<false>(wr, kc, hi);

    v8f bias;                                  // b_hh in D layout (row-dependent)
#pragma unroll
    for (int r = 0; r < 8; ++r) bias[r] = b_hh[row_base + 8 * hi + r];

    float hreg[3] = {0.f, 0.f, 0.f};           // persistently owned h values

    // prefill LDS h with h0 = 0 (hbuf[0] zeroed by init_state)
    for (int v = tid; v < (Bv * Dv) / 4; v += 192) {
        const int n = v >> 7, k = (v & 127) * 4;
        float4 hv = *(const float4*)(hbuf + n * Dv + k);
        __bf16* dst = &h_bf[n * HS + k];
        dst[0] = (__bf16)hv.x; dst[1] = (__bf16)hv.y;
        dst[2] = (__bf16)hv.z; dst[3] = (__bf16)hv.w;
    }
    __syncthreads();

    unsigned* cnt = bar;
    unsigned* gen = bar + 1;

#pragma unroll 1
    for (int t = 0; t < Tv; ++t) {
        // ---- gh^T = W_tile x h^T  (A from regs, B from LDS) -------------
        const __bf16* hrow = &h_bf[m * HS + hi * 16];
        v8f acc0 = bias;
        v8f acc1;
#pragma unroll
        for (int r = 0; r < 8; ++r) acc1[r] = 0.f;
#pragma unroll
        for (int kc = 0; kc < 16; kc += 2) {
            v16bf b0 = load_b_frag_lds(hrow + (kc + 0) * 32);
            v16bf b1 = load_b_frag_lds(hrow + (kc + 1) * 32);
            acc0 = wmma_bf16(aw[kc + 0], b0, acc0);
            acc1 = wmma_bf16(aw[kc + 1], b1, acc1);
        }
        v8f acc = acc0 + acc1;
        // D: lane col = batch m, VGPR r -> unit_local = hf*16 + 8*hi + r
#pragma unroll
        for (int r = 0; r < 8; ++r)
            gh_s[g][hf * 16 + 8 * hi + r][m] = acc[r];
        __syncthreads();

        // ---- gates + state update (512 outputs over 192 threads) --------
        const int wb = (t & 1) ^ 1;            // write buffer this step
        float* hw = hbuf + wb * (Bv * Dv);
#pragma unroll
        for (int s = 0; s < 3; ++s) {
            const int f = tid + 192 * s;
            if (f < 512) {
                const int n  = f >> 5;          // batch
                const int j  = f & 31;          // local unit
                const int ug = blockIdx.x * 32 + j;
                const size_t xb = ((size_t)n * Tv + t) * G3 + ug;
                const float xr_ = xp[xb];
                const float xz_ = xp[xb + 512];
                const float xn_ = xp[xb + 1024];
                const float rr = 1.f / (1.f + __expf(-(xr_ + gh_s[0][j][n])));
                const float zz = 1.f / (1.f + __expf(-(xz_ + gh_s[1][j][n])));
                const float nn = tanhf(xn_ + rr * gh_s[2][j][n]);
                const float hn = (1.f - zz) * nn + zz * hreg[s];
                hreg[s] = hn;
                y[((size_t)n * Tv + t) * Dv + ug] = hn;
                hw[n * Dv + ug] = hn;
                if (t + 1 < Tv) __builtin_prefetch(xp + xb + G3, 0, 0);
            }
        }
        __threadfence();                        // release our h_new slice
        __syncthreads();

        // ---- one device-scope barrier per step (16 WGs) -----------------
        if (tid == 0) {
            unsigned old = __hip_atomic_fetch_add(cnt, 1u, __ATOMIC_ACQ_REL,
                                                  __HIP_MEMORY_SCOPE_AGENT);
            if (old == 15u) {
                __hip_atomic_store(cnt, 0u, __ATOMIC_RELAXED, __HIP_MEMORY_SCOPE_AGENT);
                __hip_atomic_fetch_add(gen, 1u, __ATOMIC_RELEASE,
                                       __HIP_MEMORY_SCOPE_AGENT);
            } else {
                while (__hip_atomic_load(gen, __ATOMIC_ACQUIRE,
                                         __HIP_MEMORY_SCOPE_AGENT) < (unsigned)(t + 1))
                    __builtin_amdgcn_s_sleep(2);
            }
        }
        __syncthreads();
        __threadfence();                        // acquire other WGs' slices

        // ---- refresh broadcast h (bf16) in LDS (vectorized) -------------
        const float* hr = hbuf + wb * (Bv * Dv);
        for (int v = tid; v < (Bv * Dv) / 4; v += 192) {
            const int n = v >> 7, k = (v & 127) * 4;
            float4 hv = *(const float4*)(hr + n * Dv + k);
            __bf16* dst = &h_bf[n * HS + k];
            dst[0] = (__bf16)hv.x; dst[1] = (__bf16)hv.y;
            dst[2] = (__bf16)hv.z; dst[3] = (__bf16)hv.w;
        }
        __syncthreads();
    }
}

// ---------------------------------------------------------------------------
extern "C" void kernel_launch(void* const* d_in, const int* in_sizes, int n_in,
                              void* d_out, int out_size, void* d_ws, size_t ws_size,
                              hipStream_t stream) {
    (void)in_sizes; (void)n_in; (void)out_size; (void)ws_size;
    const float* x    = (const float*)d_in[0];
    const float* wih1 = (const float*)d_in[1];
    const float* whh1 = (const float*)d_in[2];
    const float* bih1 = (const float*)d_in[3];
    const float* bhh1 = (const float*)d_in[4];
    const float* wih2 = (const float*)d_in[5];
    const float* whh2 = (const float*)d_in[6];
    const float* bih2 = (const float*)d_in[7];
    const float* bhh2 = (const float*)d_in[8];
    float* y_out = (float*)d_out;   // holds y1 temporarily, then final y2

    const size_t xp_elems = (size_t)Bv * Tv * G3;          // 100.6M floats
    float*    xp   = (float*)d_ws;                          // reused by both layers
    float*    hbuf = xp + xp_elems;                         // 2 x [16,512] ping-pong
    unsigned* bar  = (unsigned*)(hbuf + 2 * Bv * Dv);       // {count, gen, pad}

    const int Rblocks = (Bv * Tv) / 16;                     // 4096

    // Layer 1
    xproj_gemm<false><<<Rblocks, 256, 0, stream>>>(x, wih1, bih1, xp);
    init_state<<<64, 256, 0, stream>>>(hbuf, bar);
    gru_scan<<<16, 192, 0, stream>>>(xp, whh1, bhh1, y_out, hbuf, bar);

    // Layer 2: exact-erf GELU fused into the input-projection A loads.
    // (y1 in d_out is dead once xp2 is built, so layer 2 may overwrite d_out.)
    xproj_gemm<true><<<Rblocks, 256, 0, stream>>>(y_out, wih2, bih2, xp);
    init_state<<<64, 256, 0, stream>>>(hbuf, bar);
    gru_scan<<<16, 192, 0, stream>>>(xp, whh2, bhh2, y_out, hbuf, bar);
}
